// PointNetPlus_3204045603270
// MI455X (gfx1250) — matrix operand order repaired and checked by reading, not compiled, not run on hardware
//
#include <hip/hip_runtime.h>
#include <hip/hip_bf16.h>
#include <math.h>

// ---------------------------------------------------------------------------
// PointNet++ forward for MI455X (gfx1250, wave32).
// All channel-mixing GEMMs run on v_wmma_f32_16x16x32_bf16 (f32 accumulate).
// Weights are staged TRANSPOSED (N-major) in LDS so both A and B fragments
// collapse to ds_load_b128 pairs. BN affine folded into scale/shift epilogue.
// Grouped SA MLPs fused in LDS (3 layers + maxpool per workgroup).
// ---------------------------------------------------------------------------

typedef __attribute__((ext_vector_type(16))) __bf16 v16bf;
typedef __attribute__((ext_vector_type(8)))  float  v8f;
typedef unsigned short ushort_t;

#define EPS_BN 1e-5f

__device__ __forceinline__ ushort_t f2bf(float f) {
  unsigned int u = __float_as_uint(f);
  unsigned int r = u + 0x7FFFu + ((u >> 16) & 1u);   // round-nearest-even
  return (ushort_t)(r >> 16);
}
__device__ __forceinline__ float bf2f(ushort_t h) {
  return __uint_as_float(((unsigned int)h) << 16);
}
__device__ __forceinline__ __bf16 bfbits(ushort_t h) {
  return __builtin_bit_cast(__bf16, h);
}
__device__ __forceinline__ v8f wmma_bf16(v16bf a, v16bf b, v8f c) {
  return __builtin_amdgcn_wmma_f32_16x16x32_bf16(
      /*neg_a=*/false, a, /*neg_b=*/false, b,
      /*c_mod=*/(short)0, c, /*reuse_a=*/false, /*reuse_b=*/false);
}

// ---------------------------------------------------------------------------
struct DevLayer {
  const ushort_t* W;      // Kp x Cout, row-major, bf16 bits (K zero-padded)
  const float*    scale;  // g / sqrt(1+eps)
  const float*    shift;  // b*scale + beta
  int Kp;
  int Cout;
};

// ---------------------------------------------------------------------------
// Weight prep: f32 W -> bf16 padded; fold BN into scale/shift
// ---------------------------------------------------------------------------
__global__ void k_prep(const float* __restrict__ W, const float* __restrict__ b,
                       const float* __restrict__ g, const float* __restrict__ be,
                       int Kin, int Kp, int Cout, ushort_t* __restrict__ Wp,
                       float* __restrict__ scale, float* __restrict__ shift) {
  int i = blockIdx.x * blockDim.x + threadIdx.x;
  if (i < Cout) {
    float inv = 1.0f / sqrtf(1.0f + EPS_BN);
    float s = g[i] * inv;
    scale[i] = s;
    shift[i] = b[i] * s + be[i];
  }
  int total = Kp * Cout;
  if (i < total) {
    int k = i / Cout;
    int c = i - k * Cout;
    Wp[i] = (k < Kin) ? f2bf(W[(size_t)k * Cout + c]) : (ushort_t)0;
  }
}

// ---------------------------------------------------------------------------
// Farthest point sampling: one block per batch; dist[] in LDS.
// ---------------------------------------------------------------------------
__global__ void __launch_bounds__(256)
k_fps(const float* __restrict__ xyz, int N, int S,
      int* __restrict__ fidx, float* __restrict__ newxyz) {
  __shared__ float dist[8192];
  __shared__ float rv[256];
  __shared__ int   ri[256];
  __shared__ int   s_far;
  int b = blockIdx.x, tid = threadIdx.x;
  const float* X = xyz + (size_t)b * N * 3;
  for (int i = tid; i < N; i += 256) dist[i] = 1.0e10f;
  if (tid == 0) s_far = 0;
  __syncthreads();
  for (int it = 0; it < S; ++it) {
    int far = s_far;
    float cx = X[far * 3 + 0], cy = X[far * 3 + 1], cz = X[far * 3 + 2];
    if (tid == 0) {
      fidx[(size_t)b * S + it] = far;
      newxyz[((size_t)b * S + it) * 3 + 0] = cx;
      newxyz[((size_t)b * S + it) * 3 + 1] = cy;
      newxyz[((size_t)b * S + it) * 3 + 2] = cz;
    }
    float bv = -1.0f; int bi = N;
    for (int i = tid; i < N; i += 256) {
      float dx = X[i * 3 + 0] - cx, dy = X[i * 3 + 1] - cy, dz = X[i * 3 + 2] - cz;
      float d = dx * dx + dy * dy + dz * dz;
      float dm = fminf(dist[i], d);
      dist[i] = dm;
      if (dm > bv || (dm == bv && i < bi)) { bv = dm; bi = i; }
    }
    rv[tid] = bv; ri[tid] = bi;
    __syncthreads();
    for (int o = 128; o > 0; o >>= 1) {
      if (tid < o) {
        if (rv[tid + o] > rv[tid] ||
            (rv[tid + o] == rv[tid] && ri[tid + o] < ri[tid])) {
          rv[tid] = rv[tid + o]; ri[tid] = ri[tid + o];
        }
      }
      __syncthreads();
    }
    if (tid == 0) s_far = ri[0];
    __syncthreads();
  }
}

// ---------------------------------------------------------------------------
// Ball query: one thread per centroid, deterministic in-order scan.
// ---------------------------------------------------------------------------
__global__ void k_ballquery(const float* __restrict__ xyz,
                            const float* __restrict__ newxyz,
                            int N, int S, float r2,
                            int* __restrict__ gidx, int total) {
  int t = blockIdx.x * blockDim.x + threadIdx.x;
  if (t >= total) return;
  int b = t / S;
  const float* X = xyz + (size_t)b * N * 3;
  float cx = newxyz[(size_t)t * 3 + 0];
  float cy = newxyz[(size_t)t * 3 + 1];
  float cz = newxyz[(size_t)t * 3 + 2];
  int* out = gidx + (size_t)t * 64;
  int cnt = 0, first = 0;
  for (int i = 0; i < N && cnt < 64; ++i) {
    float dx = X[i * 3 + 0] - cx, dy = X[i * 3 + 1] - cy, dz = X[i * 3 + 2] - cz;
    float d = dx * dx + dy * dy + dz * dz;
    if (d <= r2) { if (cnt == 0) first = i; out[cnt++] = i; }
  }
  if (cnt == 0) out[cnt++] = 0;
  for (; cnt < 64; ++cnt) out[cnt] = first;
}

// ---------------------------------------------------------------------------
// Fused grouped MLP: one block per (b,s) group of 64 neighbors.
// Activations ping-pong between two LDS buffers; each layer's weights are
// staged once into LDS, TRANSPOSED (n-major), so B-fragments are two
// ds_load_b128 per WMMA, same as A-fragments.
// ---------------------------------------------------------------------------
struct GroupArgs {
  const float*    xyz;       // B x N x 3
  const ushort_t* feats;     // B x N x Cf bf16 (or null)
  const int*      gidx;      // (B*S) x 64
  const float*    centers;   // (B*S) x 3
  ushort_t*       outFeats;  // (B*S) x Cout_last bf16
  int N, S, Cf, Kp0, nL, maxdim, maxW;
  DevLayer L[3];
};

__global__ void __launch_bounds__(256) k_group_mlp(GroupArgs ga) {
  extern __shared__ unsigned char smem_raw[];
  int stride = ga.maxdim + 8;
  ushort_t* sA   = (ushort_t*)smem_raw;
  ushort_t* sB   = sA + 64 * stride;
  ushort_t* sW   = sB + 64 * stride;           // transposed weights: Cout x (Kin+8)
  int*      sIdx = (int*)(sW + ga.maxW);
  float*    sCtr = (float*)(sIdx + 64);

  int g = blockIdx.x;
  int b = g / ga.S;
  int tid = threadIdx.x;
  int lane = tid & 31, wave = tid >> 5;
  int half = lane >> 4, lm = lane & 15;

  if (tid < 64) sIdx[tid] = ga.gidx[(size_t)g * 64 + tid];
  if (tid < 3)  sCtr[tid] = ga.centers[(size_t)g * 3 + tid];
  __syncthreads();

  int Kp0 = ga.Kp0;
  for (int e = tid; e < 64 * Kp0; e += 256) {
    int j = e / Kp0, c = e - j * Kp0;
    int gi = sIdx[j];
    float v = 0.0f;
    if (c < 3)              v = ga.xyz[((size_t)b * ga.N + gi) * 3 + c] - sCtr[c];
    else if (c < 3 + ga.Cf) v = bf2f(ga.feats[((size_t)b * ga.N + gi) * ga.Cf + (c - 3)]);
    sA[j * stride + c] = f2bf(v);
  }
  __syncthreads();

  ushort_t* cur = sA; ushort_t* nxt = sB;
  int Kin = Kp0;
  for (int l = 0; l < ga.nL; ++l) {
    const ushort_t* Wg = ga.L[l].W;
    const float* sc = ga.L[l].scale;
    const float* sh = ga.L[l].shift;
    int Cout = ga.L[l].Cout;
    int sws  = Kin + 8;                          // transposed row stride (16B aligned)

    // Stage layer weights transposed: sW[n*sws + k] = W[k*Cout + n]
    int halfc = Cout >> 1;
    for (int e = tid; e < Kin * halfc; e += 256) {
      int k = e / halfc, n2 = (e - k * halfc) * 2;
      unsigned int v = *(const unsigned int*)(Wg + (size_t)k * Cout + n2);
      sW[(size_t)n2 * sws + k]       = (ushort_t)(v & 0xFFFFu);
      sW[(size_t)(n2 + 1) * sws + k] = (ushort_t)(v >> 16);
    }
    __syncthreads();

    int nTiles = (Cout >> 4) * 4;                // 4 row tiles x Cout/16 col tiles
    for (int t = wave; t < nTiles; t += 8) {
      int tm = t & 3, tn = t >> 2;
      int n = tn * 16 + lm;
      v8f acc = {};
      for (int k0 = 0; k0 < Kin; k0 += 32) {
        v16bf a;
        #pragma unroll
        for (int j = 0; j < 16; ++j) {
          int k = k0 + half * 16 + j;   // A 16x32 layout
          a[j] = bfbits(cur[(tm * 16 + lm) * stride + k]);
        }
        v16bf bv;
        #pragma unroll
        for (int j = 0; j < 16; ++j) {                          // B 32x16 layout
          bv[j] = bfbits(sW[(size_t)n * sws + k0 + half * 16 + j]);
        }
        acc = wmma_bf16(a, bv, acc);
      }
      float s0 = sc[n], s1 = sh[n];
      #pragma unroll
      for (int r = 0; r < 8; ++r) {
        int m = tm * 16 + half * 8 + r;                         // D layout
        float y = fmaxf(acc[r] * s0 + s1, 0.0f);
        nxt[m * stride + n] = f2bf(y);
      }
    }
    __syncthreads();
    ushort_t* tmp = cur; cur = nxt; nxt = tmp;
    Kin = Cout;
  }

  for (int c = tid; c < Kin; c += 256) {                        // maxpool over 64 pts
    float mx = -3.0e38f;
    for (int j = 0; j < 64; ++j) mx = fmaxf(mx, bf2f(cur[j * stride + c]));
    ga.outFeats[(size_t)g * Kin + c] = f2bf(mx);
  }
}

// ---------------------------------------------------------------------------
// Generic pointwise GEMM: Y[MxN] = act((X[MxKp] @ W[KpxN]) * scale + shift)
// 64x64 block, 8 waves x 2 (16x16) tiles. X staged with uint4 copies;
// W staged transposed so B-fragments vectorize to ds_load_b128.
// outT != null => write f32 transposed (B,N,npts) layout (final FC).
// ---------------------------------------------------------------------------
__global__ void __launch_bounds__(256)
k_gemm(const ushort_t* __restrict__ X, const ushort_t* __restrict__ W,
       const float* __restrict__ scale, const float* __restrict__ shift,
       int M, int N, int Kp, ushort_t* __restrict__ Y,
       float* __restrict__ outT, int relu, int npts) {
  __shared__ ushort_t Xs[64 * 40];     // 64 rows x 32 k (+8 pad)
  __shared__ ushort_t Wls[64 * 40];    // transposed: 64 n-rows x 32 k (+8 pad)
  int tid = threadIdx.x;
  int lane = tid & 31, wave = tid >> 5;
  int half = lane >> 4, lm = lane & 15;
  int m0 = blockIdx.y * 64, n0 = blockIdx.x * 64;
  int tm = wave & 3, tn0 = wave >> 2, tn1 = tn0 + 2;
  v8f acc0 = {}, acc1 = {};

  for (int k0 = 0; k0 < Kp; k0 += 32) {
    __syncthreads();
    {   // X tile: one 16-byte copy per thread (rows are 64B-aligned)
      int r = tid >> 2, c = tid & 3;
      *(uint4*)(Xs + r * 40 + c * 8) =
          *(const uint4*)(X + (size_t)(m0 + r) * Kp + k0 + c * 8);
    }
    {   // W tile transposed: coalesced uint reads, scattered LDS writes
      for (int e = tid; e < 32 * 32; e += 256) {
        int k = e >> 5, n2 = (e & 31) * 2;
        unsigned int v = *(const unsigned int*)(W + (size_t)(k0 + k) * N + n0 + n2);
        Wls[(n2 + 0) * 40 + k] = (ushort_t)(v & 0xFFFFu);
        Wls[(n2 + 1) * 40 + k] = (ushort_t)(v >> 16);
      }
    }
    if (k0 + 32 < Kp) {
      __builtin_prefetch(W + (size_t)(k0 + 32) * N + n0, 0, 1);   // global_prefetch
      __builtin_prefetch(X + (size_t)(m0 + (tid >> 2)) * Kp + k0 + 32, 0, 1);
    }
    __syncthreads();
    v16bf a;
    #pragma unroll
    for (int j = 0; j < 16; ++j) {
      int k = half * 16 + j;
      a[j] = bfbits(Xs[(tm * 16 + lm) * 40 + k]);
    }
    v16bf b0, b1;
    #pragma unroll
    for (int j = 0; j < 16; ++j) {
      b0[j] = bfbits(Wls[(tn0 * 16 + lm) * 40 + half * 16 + j]);
      b1[j] = bfbits(Wls[(tn1 * 16 + lm) * 40 + half * 16 + j]);
    }
    acc0 = wmma_bf16(a, b0, acc0);
    acc1 = wmma_bf16(a, b1, acc1);
  }

  int gm = m0 + tm * 16 + half * 8;
  #pragma unroll 2
  for (int pass = 0; pass < 2; ++pass) {
    v8f acc = pass ? acc1 : acc0;
    int n = n0 + (pass ? tn1 : tn0) * 16 + lm;
    float sc = scale[n], sh = shift[n];
    #pragma unroll
    for (int r = 0; r < 8; ++r) {
      float y = acc[r] * sc + sh;
      if (relu) y = fmaxf(y, 0.0f);
      int m = gm + r;
      if (outT) {
        int bb = m / npts, nn = m - bb * npts;
        outT[((size_t)bb * N + n) * npts + nn] = y;   // (B, C, npts)
      } else {
        Y[(size_t)m * N + n] = f2bf(y);
      }
    }
  }
}

// ---------------------------------------------------------------------------
// Small glue kernels
// ---------------------------------------------------------------------------
__global__ void k_build_sa3(const float* __restrict__ l2xyz,
                            const ushort_t* __restrict__ l2p,
                            ushort_t* __restrict__ X, int rows) {
  int i = blockIdx.x * blockDim.x + threadIdx.x;
  if (i >= rows * 288) return;
  int r = i / 288, c = i - r * 288;
  float v = 0.0f;
  if (c < 3)        v = l2xyz[(size_t)r * 3 + c];
  else if (c < 259) v = bf2f(l2p[(size_t)r * 256 + (c - 3)]);
  X[i] = f2bf(v);
}

__global__ void k_maxpool_rows(const ushort_t* __restrict__ Y,
                               ushort_t* __restrict__ out,
                               int rows, int C, int total) {
  int t = blockIdx.x * blockDim.x + threadIdx.x;
  if (t >= total) return;
  int b = t / C, c = t - b * C;
  const ushort_t* base = Y + (size_t)b * rows * C + c;
  float mx = -3.0e38f;
  for (int r = 0; r < rows; ++r) mx = fmaxf(mx, bf2f(base[(size_t)r * C]));
  out[(size_t)b * C + c] = f2bf(mx);
}

__global__ void k_build_fp3(const ushort_t* __restrict__ l2p,
                            const ushort_t* __restrict__ l3p,
                            ushort_t* __restrict__ X, int rows) {
  int i = blockIdx.x * blockDim.x + threadIdx.x;
  if (i >= rows * 1280) return;
  int r = i / 1280, c = i - r * 1280;
  int b = r >> 7;  // / 128
  X[i] = (c < 256) ? l2p[(size_t)r * 256 + c]
                   : l3p[(size_t)b * 1024 + (c - 256)];
}

// fp2: 3-NN interp of l2p2 (256ch over 128 pts) onto 512 pts + concat l1p(128)
__global__ void k_interp_fp2(const float* __restrict__ l1xyz,
                             const float* __restrict__ l2xyz,
                             const ushort_t* __restrict__ l1p,
                             const ushort_t* __restrict__ l2p2,
                             ushort_t* __restrict__ X) {
  int r = blockIdx.x * blockDim.x + threadIdx.x;
  if (r >= 16 * 512) return;
  int b = r >> 9;
  float px = l1xyz[(size_t)r * 3 + 0], py = l1xyz[(size_t)r * 3 + 1], pz = l1xyz[(size_t)r * 3 + 2];
  const float* q = l2xyz + (size_t)b * 128 * 3;
  float d0 = 3e38f, d1 = 3e38f, d2 = 3e38f; int i0 = 0, i1 = 0, i2 = 0;
  for (int j = 0; j < 128; ++j) {
    float dx = px - q[j * 3], dy = py - q[j * 3 + 1], dz = pz - q[j * 3 + 2];
    float d = dx * dx + dy * dy + dz * dz;
    if (d < d0)      { d2 = d1; i2 = i1; d1 = d0; i1 = i0; d0 = d; i0 = j; }
    else if (d < d1) { d2 = d1; i2 = i1; d1 = d; i1 = j; }
    else if (d < d2) { d2 = d; i2 = j; }
  }
  float w0 = 1.0f / (d0 + 1e-8f), w1 = 1.0f / (d1 + 1e-8f), w2 = 1.0f / (d2 + 1e-8f);
  float ws = w0 + w1 + w2; w0 /= ws; w1 /= ws; w2 /= ws;
  ushort_t* xr = X + (size_t)r * 384;
  const ushort_t* f = l1p + (size_t)r * 128;
  for (int c = 0; c < 128; ++c) xr[c] = f[c];
  const ushort_t* p0 = l2p2 + ((size_t)b * 128 + i0) * 256;
  const ushort_t* p1 = l2p2 + ((size_t)b * 128 + i1) * 256;
  const ushort_t* p2 = l2p2 + ((size_t)b * 128 + i2) * 256;
  for (int c = 0; c < 256; ++c)
    xr[128 + c] = f2bf(w0 * bf2f(p0[c]) + w1 * bf2f(p1[c]) + w2 * bf2f(p2[c]));
}

// fp1: 3-NN interp of l1p2 (128ch over 512 pts) onto 8192 pts + concat l0xyz(3); pad to 160
__global__ void k_interp_fp1(const float* __restrict__ l0xyz,
                             const float* __restrict__ l1xyz,
                             const ushort_t* __restrict__ l1p2,
                             ushort_t* __restrict__ X) {
  int r = blockIdx.x * blockDim.x + threadIdx.x;
  if (r >= 16 * 8192) return;
  int b = r >> 13;
  float px = l0xyz[(size_t)r * 3 + 0], py = l0xyz[(size_t)r * 3 + 1], pz = l0xyz[(size_t)r * 3 + 2];
  const float* q = l1xyz + (size_t)b * 512 * 3;
  float d0 = 3e38f, d1 = 3e38f, d2 = 3e38f; int i0 = 0, i1 = 0, i2 = 0;
  for (int j = 0; j < 512; ++j) {
    float dx = px - q[j * 3], dy = py - q[j * 3 + 1], dz = pz - q[j * 3 + 2];
    float d = dx * dx + dy * dy + dz * dz;
    if (d < d0)      { d2 = d1; i2 = i1; d1 = d0; i1 = i0; d0 = d; i0 = j; }
    else if (d < d1) { d2 = d1; i2 = i1; d1 = d; i1 = j; }
    else if (d < d2) { d2 = d; i2 = j; }
  }
  float w0 = 1.0f / (d0 + 1e-8f), w1 = 1.0f / (d1 + 1e-8f), w2 = 1.0f / (d2 + 1e-8f);
  float ws = w0 + w1 + w2; w0 /= ws; w1 /= ws; w2 /= ws;
  ushort_t* xr = X + (size_t)r * 160;
  xr[0] = f2bf(px); xr[1] = f2bf(py); xr[2] = f2bf(pz);
  const ushort_t* p0 = l1p2 + ((size_t)b * 512 + i0) * 128;
  const ushort_t* p1 = l1p2 + ((size_t)b * 512 + i1) * 128;
  const ushort_t* p2 = l1p2 + ((size_t)b * 512 + i2) * 128;
  for (int c = 0; c < 128; ++c)
    xr[3 + c] = f2bf(w0 * bf2f(p0[c]) + w1 * bf2f(p1[c]) + w2 * bf2f(p2[c]));
  for (int c = 131; c < 160; ++c) xr[c] = 0;   // zero K padding
}

// ---------------------------------------------------------------------------
// Host orchestration
// ---------------------------------------------------------------------------
extern "C" void kernel_launch(void* const* d_in, const int* in_sizes, int n_in,
                              void* d_out, int out_size, void* d_ws, size_t ws_size,
                              hipStream_t stream) {
  (void)in_sizes; (void)n_in; (void)out_size; (void)ws_size;
  const int B = 16, N0 = 8192, S1 = 512, S2 = 128;
  const float* l0xyz = (const float*)d_in[0];

  // JAX tree-flatten order (alphabetical dict keys; tuples (W,b,g,beta)):
  // fc:1..4 | fp1:5..16 | fp2:17..24 | fp3:25..32 | sa1:33..44 | sa2:45..56 | sa3:57..68
  const int baseIdx[17] = {33,37,41, 45,49,53, 57,61,65, 25,29, 17,21, 5,9,13, 1};
  const int kinArr[17]  = { 3,64,64, 131,128,128, 259,256,512, 1280,256, 384,256, 131,128,128, 128};
  const int coutArr[17] = {64,64,128,128,128,256, 256,512,1024, 256,256, 256,128, 128,128,128, 128};

  char* wsb = (char*)d_ws;
  size_t off = 0;
  auto alloc = [&](size_t bytes) -> void* {
    off = (off + 255) & ~(size_t)255;
    void* p = wsb + off;
    off += bytes;
    return p;
  };

  DevLayer hl[17];
  for (int i = 0; i < 17; ++i) {
    int Kp = (kinArr[i] + 31) & ~31;
    int Cout = coutArr[i];
    ushort_t* wp = (ushort_t*)alloc((size_t)Kp * Cout * 2);
    float* sc = (float*)alloc((size_t)Cout * 4);
    float* sh = (float*)alloc((size_t)Cout * 4);
    hl[i].W = wp; hl[i].scale = sc; hl[i].shift = sh; hl[i].Kp = Kp; hl[i].Cout = Cout;
    int total = Kp * Cout;
    k_prep<<<(total + 255) / 256, 256, 0, stream>>>(
        (const float*)d_in[baseIdx[i] + 0], (const float*)d_in[baseIdx[i] + 1],
        (const float*)d_in[baseIdx[i] + 2], (const float*)d_in[baseIdx[i] + 3],
        kinArr[i], Kp, Cout, wp, sc, sh);
  }

  int*      fidx1 = (int*)alloc((size_t)B * S1 * 4);
  float*    l1xyz = (float*)alloc((size_t)B * S1 * 3 * 4);
  int*      gidx1 = (int*)alloc((size_t)B * S1 * 64 * 4);
  ushort_t* l1p   = (ushort_t*)alloc((size_t)B * S1 * 128 * 2);
  int*      fidx2 = (int*)alloc((size_t)B * S2 * 4);
  float*    l2xyz = (float*)alloc((size_t)B * S2 * 3 * 4);
  int*      gidx2 = (int*)alloc((size_t)B * S2 * 64 * 4);
  ushort_t* l2p   = (ushort_t*)alloc((size_t)B * S2 * 256 * 2);
  ushort_t* Xsa3  = (ushort_t*)alloc((size_t)B * S2 * 288 * 2);
  ushort_t* Y3a   = (ushort_t*)alloc((size_t)B * S2 * 256 * 2);
  ushort_t* Y3b   = (ushort_t*)alloc((size_t)B * S2 * 512 * 2);
  ushort_t* Y3c   = (ushort_t*)alloc((size_t)B * S2 * 1024 * 2);
  ushort_t* l3p   = (ushort_t*)alloc((size_t)B * 1024 * 2);
  ushort_t* Xfp3  = (ushort_t*)alloc((size_t)B * S2 * 1280 * 2);
  ushort_t* Yfp3  = (ushort_t*)alloc((size_t)B * S2 * 256 * 2);
  ushort_t* l2p2  = (ushort_t*)alloc((size_t)B * S2 * 256 * 2);
  ushort_t* Xfp2  = (ushort_t*)alloc((size_t)B * S1 * 384 * 2);
  ushort_t* Yfp2  = (ushort_t*)alloc((size_t)B * S1 * 256 * 2);
  ushort_t* l1p2  = (ushort_t*)alloc((size_t)B * S1 * 128 * 2);
  ushort_t* Xfp1  = (ushort_t*)alloc((size_t)B * N0 * 160 * 2);
  ushort_t* Yfp1a = (ushort_t*)alloc((size_t)B * N0 * 128 * 2);
  ushort_t* Yfp1b = (ushort_t*)alloc((size_t)B * N0 * 128 * 2);

  auto gemm = [&](const ushort_t* Xp, const DevLayer& L, int M, ushort_t* Yp,
                  int relu, float* outT, int npts) {
    dim3 grid(L.Cout / 64, M / 64);
    k_gemm<<<grid, 256, 0, stream>>>(Xp, L.W, L.scale, L.shift, M, L.Cout, L.Kp,
                                     Yp, outT, relu, npts);
  };
  auto maxWElems = [&](int a, int bI, int c) -> int {
    int m = hl[a].Cout * (hl[a].Kp + 8);
    int m2 = hl[bI].Cout * (hl[bI].Kp + 8);
    int m3 = hl[c].Cout * (hl[c].Kp + 8);
    if (m2 > m) m = m2;
    if (m3 > m) m = m3;
    return m;
  };
  auto dynBytes = [](int maxdim, int maxW) -> size_t {
    size_t stride = (size_t)maxdim + 8;
    return (2 * 64 * stride + (size_t)maxW) * 2 + 64 * 4 + 4 * 4 + 64;
  };

  // ---- SA1 -----------------------------------------------------------------
  k_fps<<<B, 256, 0, stream>>>(l0xyz, N0, S1, fidx1, l1xyz);
  k_ballquery<<<(B * S1 + 255) / 256, 256, 0, stream>>>(
      l0xyz, l1xyz, N0, S1, 0.2f * 0.2f, gidx1, B * S1);
  GroupArgs ga1{};
  ga1.xyz = l0xyz; ga1.feats = nullptr; ga1.gidx = gidx1; ga1.centers = l1xyz;
  ga1.outFeats = l1p; ga1.N = N0; ga1.S = S1; ga1.Cf = 0; ga1.Kp0 = 32;
  ga1.nL = 3; ga1.maxdim = 128; ga1.maxW = maxWElems(0, 1, 2);
  ga1.L[0] = hl[0]; ga1.L[1] = hl[1]; ga1.L[2] = hl[2];
  k_group_mlp<<<B * S1, 256, dynBytes(128, ga1.maxW), stream>>>(ga1);

  // ---- SA2 -----------------------------------------------------------------
  k_fps<<<B, 256, 0, stream>>>(l1xyz, S1, S2, fidx2, l2xyz);
  k_ballquery<<<(B * S2 + 255) / 256, 256, 0, stream>>>(
      l1xyz, l2xyz, S1, S2, 0.4f * 0.4f, gidx2, B * S2);
  GroupArgs ga2{};
  ga2.xyz = l1xyz; ga2.feats = l1p; ga2.gidx = gidx2; ga2.centers = l2xyz;
  ga2.outFeats = l2p; ga2.N = S1; ga2.S = S2; ga2.Cf = 128; ga2.Kp0 = 160;
  ga2.nL = 3; ga2.maxdim = 256; ga2.maxW = maxWElems(3, 4, 5);
  ga2.L[0] = hl[3]; ga2.L[1] = hl[4]; ga2.L[2] = hl[5];
  k_group_mlp<<<B * S2, 256, dynBytes(256, ga2.maxW), stream>>>(ga2);

  // ---- SA3 (group_all): pointwise MLP on 2048 points + maxpool over 128 ----
  int Msa3 = B * S2;  // 2048
  k_build_sa3<<<(Msa3 * 288 + 255) / 256, 256, 0, stream>>>(l2xyz, l2p, Xsa3, Msa3);
  gemm(Xsa3, hl[6], Msa3, Y3a, 1, nullptr, 0);
  gemm(Y3a,  hl[7], Msa3, Y3b, 1, nullptr, 0);
  gemm(Y3b,  hl[8], Msa3, Y3c, 1, nullptr, 0);
  k_maxpool_rows<<<(B * 1024 + 255) / 256, 256, 0, stream>>>(Y3c, l3p, S2, 1024, B * 1024);

  // ---- FP3 (S==1 broadcast) ------------------------------------------------
  k_build_fp3<<<(Msa3 * 1280 + 255) / 256, 256, 0, stream>>>(l2p, l3p, Xfp3, Msa3);
  gemm(Xfp3, hl[9],  Msa3, Yfp3, 1, nullptr, 0);
  gemm(Yfp3, hl[10], Msa3, l2p2, 1, nullptr, 0);

  // ---- FP2 -----------------------------------------------------------------
  k_interp_fp2<<<(B * S1 + 255) / 256, 256, 0, stream>>>(l1xyz, l2xyz, l1p, l2p2, Xfp2);
  gemm(Xfp2, hl[11], B * S1, Yfp2, 1, nullptr, 0);
  gemm(Yfp2, hl[12], B * S1, l1p2, 1, nullptr, 0);

  // ---- FP1 -----------------------------------------------------------------
  k_interp_fp1<<<(B * N0 + 255) / 256, 256, 0, stream>>>(l0xyz, l1xyz, l1p2, Xfp1);
  gemm(Xfp1,  hl[13], B * N0, Yfp1a, 1, nullptr, 0);
  gemm(Yfp1a, hl[14], B * N0, Yfp1b, 1, nullptr, 0);
  gemm(Yfp1b, hl[15], B * N0, Yfp1a, 1, nullptr, 0);

  // ---- FC (no ReLU) + transpose to (B, 128, 8192) f32 into d_out -----------
  gemm(Yfp1a, hl[16], B * N0, nullptr, 0, (float*)d_out, N0);
}
